// PatchGATv2_34514357191323
// MI455X (gfx1250) — compile-verified, matrix operand order checked by reading, not run on hardware
//
#include <hip/hip_runtime.h>
#include <math.h>
#include <stdint.h>

typedef __attribute__((ext_vector_type(16))) __bf16 v16bf;
typedef __attribute__((ext_vector_type(8)))  __bf16 v8bf;
typedef __attribute__((ext_vector_type(8)))  float  v8f;
typedef __attribute__((__vector_size__(16))) int    v4i_t;

#define AS1 __attribute__((address_space(1)))
#define AS3 __attribute__((address_space(3)))

#if defined(__has_builtin)
#  if __has_builtin(__builtin_amdgcn_global_load_async_to_lds_b128)
#    define HAVE_ASYNC_LDS 1
#  endif
#endif
#ifndef HAVE_ASYNC_LDS
#  define HAVE_ASYNC_LDS 0
#endif

#define NNODES 20000
#define NEDGES 320000
#define NHEADS 4

static __device__ __forceinline__ __bf16 f2bf(float f) {
    unsigned u = __float_as_uint(f);
    unsigned r = (u + 0x7FFFu + ((u >> 16) & 1u)) >> 16;
    unsigned short us = (unsigned short)r;
    return __builtin_bit_cast(__bf16, us);
}

static __device__ __forceinline__ void atomicMaxF(float* addr, float val) {
    if (!(val == val)) return;  // skip NaN
    int bits = __float_as_int(val);
    if (bits >= 0) atomicMax((int*)addr, bits);
    else           atomicMin((unsigned int*)addr, (unsigned int)bits);
}

static __device__ __forceinline__ v16bf load_afrag(const __bf16* p) {
    v8bf lo = *(const v8bf*)p;
    v8bf hi = *(const v8bf*)(p + 16);
    return __builtin_shufflevector(lo, hi, 0, 1, 2, 3, 4, 5, 6, 7,
                                           8, 9, 10, 11, 12, 13, 14, 15);
}

// ---------------------------------------------------------------- elementwise
__global__ void k_cast_bf16(const float* __restrict__ in, __bf16* __restrict__ out, int n) {
    int i = blockIdx.x * blockDim.x + threadIdx.x;
    if (i < n) out[i] = f2bf(in[i]);
}

__global__ void k_fill(float* __restrict__ p, float v, int n) {
    int i = blockIdx.x * blockDim.x + threadIdx.x;
    if (i < n) p[i] = v;
}

// ---------------------------------------------------------------- B packing
// Repack W[Fin,K] (f32 row-major) into WMMA B-fragment order, bf16:
// P[ct][kf][lane][slot] with ct = 16-col tile, kf = 32-wide k step.
// Lane L holds column n=L&15; khalf=L>>4 selects K sub-groups
// {kb..kb+7, kb+16..kb+23}, kb = khalf*8 (mirrors the ISA 16-bit A layout).
__global__ void k_pack_b(const float* __restrict__ W, __bf16* __restrict__ P, int Fin, int K) {
    int idx = blockIdx.x * blockDim.x + threadIdx.x;
    if (idx >= Fin * K) return;
    int frag   = idx >> 9;
    int within = idx & 511;
    int lane = within >> 4, slot = within & 15;
    int kfrags = Fin >> 5;
    int ct = frag / kfrags, kf = frag - ct * kfrags;
    int n = lane & 15, khalf = lane >> 4;
    int klocal = khalf * 8 + (slot & 7) + ((slot >> 3) << 4);
    int k = kf * 32 + klocal;
    P[idx] = f2bf(W[(size_t)k * K + ct * 16 + n]);
}

// ---------------------------------------------------------------- WMMA GEMM
// C[N,K] = A[N,Fin](bf16) * W[Fin,K] with W pre-packed into fragments (P).
// Block = 256 threads = 8 waves = 4 row-groups x 2 col tiles; each wave owns
// 64 rows x 16 cols (4 accumulator tiles -> 4 WMMAs per B fragment).
// The packed-B panel for the block's two column tiles is staged in LDS once
// via the gfx1250 async-LDS copy (ASYNCcnt) and reused by every row tile.
// Interleaved load/WMMA ordering: the scheduler pipelines next-tile loads
// under the current WMMA and 2x-unrolls the k-loop (verified in disasm).
__global__ void __launch_bounds__(256)
k_gemm_wmma(const __bf16* __restrict__ A, const __bf16* __restrict__ P,
            float* __restrict__ C, int N, int Fin, int K) {
    __shared__ __align__(16) __bf16 Bs[16384];   // up to 32 KB panel
    const int lane  = threadIdx.x & 31;
    const int wave  = threadIdx.x >> 5;
    const int rg    = wave & 3;                  // row group 0..3
    const int ctl   = wave >> 2;                 // local col tile 0..1
    const int nidx  = lane & 15;
    const int khalf = lane >> 4;
    const int kfrags = Fin >> 5;
    const int rowTiles = N >> 4;

    // ---- stage packed-B panel (2 col tiles x kfrags x 512 bf16) into LDS
    const int ctBase = blockIdx.y * 2;
    const int panelElems = 2 * kfrags * 512;
    const __bf16* psrc = P + (size_t)ctBase * kfrags * 512;
#if HAVE_ASYNC_LDS
    {
        const int chunks = panelElems >> 3;      // 16-byte chunks
        for (int i = threadIdx.x; i < chunks; i += 256) {
            AS1 v4i_t* g = (AS1 v4i_t*)(unsigned long long)(uintptr_t)(psrc + (size_t)i * 8);
            AS3 v4i_t* l = (AS3 v4i_t*)(unsigned)(uintptr_t)(Bs + (size_t)i * 8);
            __builtin_amdgcn_global_load_async_to_lds_b128(g, l, 0, 0);
        }
#  if __has_builtin(__builtin_amdgcn_s_wait_asynccnt)
        __builtin_amdgcn_s_wait_asynccnt(0);
#  else
        asm volatile("s_wait_asynccnt 0" ::: "memory");
#  endif
    }
#else
    {
        const int chunks = panelElems >> 3;
        const uint4* s4 = (const uint4*)psrc;
        uint4* d4 = (uint4*)Bs;
        for (int i = threadIdx.x; i < chunks; i += 256) d4[i] = s4[i];
    }
#endif
    __syncthreads();

    // ---- per-wave tiles (clamped so EXEC stays all-1s for WMMA)
    int rt[4]; bool em[4];
    #pragma unroll
    for (int j = 0; j < 4; ++j) {
        int t = blockIdx.x * 16 + rg * 4 + j;
        em[j] = (t < rowTiles);
        rt[j] = em[j] ? t : (rowTiles - 1);
    }
    const int col0 = (ctBase + ctl) * 16;

    const __bf16* arow[4];
    #pragma unroll
    for (int j = 0; j < 4; ++j)
        arow[j] = A + (size_t)(rt[j] * 16 + nidx) * Fin + khalf * 8;
    const __bf16* bbase = Bs + (size_t)ctl * kfrags * 512 + lane * 16;

    v8f acc[4] = {};
    for (int kf = 0; kf < kfrags; ++kf) {
        const v16bf b = *(const v16bf*)(bbase + (size_t)kf * 512);   // ds_load_b128 x2
        if (kf + 1 < kfrags)
            __builtin_prefetch(arow[0] + (size_t)(kf + 1) * 32, 0, 0);  // global_prefetch_b8
        #pragma unroll
        for (int j = 0; j < 4; ++j) {
            v16bf a = load_afrag(arow[j] + (size_t)kf * 32);         // global_load_b128 x2
            acc[j] = __builtin_amdgcn_wmma_f32_16x16x32_bf16(
                false, a, false, b, (short)0, acc[j], false, false);
        }
    }

    #pragma unroll
    for (int j = 0; j < 4; ++j) {
        if (em[j]) {
            const int mb = rt[j] * 16 + khalf * 8;
            #pragma unroll
            for (int v = 0; v < 8; ++v)
                C[(size_t)(mb + v) * K + col0 + nidx] = acc[j][v];
        }
    }
}

// ---------------------------------------------------------------- edge kernels
__global__ void k_edge_logit_max(const float* __restrict__ fs, const float* __restrict__ fd,
                                 const int* __restrict__ src, const int* __restrict__ dst,
                                 const float* __restrict__ attn,
                                 float* __restrict__ logit, float* __restrict__ m,
                                 int E, int H, int D) {
    int idx = blockIdx.x * blockDim.x + threadIdx.x;
    if (idx >= E * H) return;
    int e = idx / H, h = idx - e * H;
    int s = src[e], d = dst[e];
    int K = H * D;
    const float4* ps = (const float4*)(fs + (size_t)s * K + h * D);
    const float4* pd = (const float4*)(fd + (size_t)d * K + h * D);
    const float4* pa = (const float4*)(attn + h * D);
    float acc = 0.f;
    for (int k = 0; k < (D >> 2); ++k) {
        float4 a = ps[k], b = pd[k], w = pa[k];
        float v0 = a.x + b.x; v0 = (v0 > 0.f) ? v0 : 0.2f * v0;
        float v1 = a.y + b.y; v1 = (v1 > 0.f) ? v1 : 0.2f * v1;
        float v2 = a.z + b.z; v2 = (v2 > 0.f) ? v2 : 0.2f * v2;
        float v3 = a.w + b.w; v3 = (v3 > 0.f) ? v3 : 0.2f * v3;
        acc = fmaf(w.x, v0, acc); acc = fmaf(w.y, v1, acc);
        acc = fmaf(w.z, v2, acc); acc = fmaf(w.w, v3, acc);
    }
    logit[idx] = acc;
    atomicMaxF(m + (size_t)d * H + h, acc);
}

__global__ void k_edge_exp_sum(float* __restrict__ logit, const int* __restrict__ dst,
                               const float* __restrict__ m, float* __restrict__ ssum,
                               int E, int H) {
    int idx = blockIdx.x * blockDim.x + threadIdx.x;
    if (idx >= E * H) return;
    int e = idx / H, h = idx - e * H;
    int d = dst[e];
    float a = __expf(logit[idx] - m[(size_t)d * H + h]);
    logit[idx] = a;
    atomicAdd(&ssum[(size_t)d * H + h], a);
}

__global__ void k_edge_accum(const float* __restrict__ a, const float* __restrict__ ssum,
                             const float* __restrict__ fs, const int* __restrict__ src,
                             const int* __restrict__ dst, float* __restrict__ acc,
                             int E, int H, int D) {
    int K = H * D;
    int idx = blockIdx.x * blockDim.x + threadIdx.x;
    if (idx >= E * K) return;
    int e = idx / K, j = idx - e * K;
    int h = j / D;
    int s = src[e], d = dst[e];
    float alpha = a[(size_t)e * H + h] / ssum[(size_t)d * H + h];
    atomicAdd(&acc[(size_t)d * K + j], alpha * fs[(size_t)s * K + j]);
}

__global__ void k_finalize(const float* __restrict__ acc, const float* __restrict__ bias,
                           float* __restrict__ hout, int N, int H, int D) {
    int idx = blockIdx.x * blockDim.x + threadIdx.x;
    if (idx >= N * D) return;
    int nn = idx / D, d = idx - nn * D;
    int K = H * D;
    float s = 0.f;
    for (int h = 0; h < H; ++h) s += acc[(size_t)nn * K + h * D + d] + bias[h * D + d];
    s *= (1.0f / H);
    hout[idx] = (s > 0.f) ? s : 0.01f * s;
}

__global__ void k_readout(const float* __restrict__ h, float* __restrict__ out, int N, int D) {
    int idx = blockIdx.x * blockDim.x + threadIdx.x;
    if (idx >= N * D) return;
    int j = idx - (idx / D) * D;
    atomicAdd(&out[j], h[idx] * (1.0f / (float)N));
}

// ---------------------------------------------------------------- orchestration
extern "C" void kernel_launch(void* const* d_in, const int* in_sizes, int n_in,
                              void* d_out, int out_size, void* d_ws, size_t ws_size,
                              hipStream_t stream) {
    (void)in_sizes; (void)n_in; (void)out_size; (void)ws_size;

    const int N = NNODES, E = NEDGES, H = NHEADS;
    const float* n_feat = (const float*)d_in[0];
    const int*   src    = (const int*)d_in[1];
    const int*   dst    = (const int*)d_in[2];
    const float* Wl[3]   = {(const float*)d_in[3], (const float*)d_in[7],  (const float*)d_in[11]};
    const float* Wr[3]   = {(const float*)d_in[4], (const float*)d_in[8],  (const float*)d_in[12]};
    const float* attn[3] = {(const float*)d_in[5], (const float*)d_in[9],  (const float*)d_in[13]};
    const float* bias[3] = {(const float*)d_in[6], (const float*)d_in[10], (const float*)d_in[14]};
    const int FinArr[3] = {512, 128, 128};
    const int DArr[3]   = {128, 128, 64};

    // workspace layout
    char* ws = (char*)d_ws;
    __bf16* Abf  = (__bf16*)ws;                                   // N*512 bf16 = 20.48 MB
    __bf16* Plbf = (__bf16*)(ws + 20971520);                      // packed Wl, <=512 KB
    __bf16* Prbf = Plbf + 512 * 512;                              // packed Wr, <=512 KB
    float* fs    = (float*)(ws + 20971520 + 2097152);             // N*512 f32 (also hnext)
    float* fd    = fs + (size_t)N * 512;                          // N*512 f32 (also acc)
    float* logit = fd + (size_t)N * 512;                          // E*H f32 (also alpha)
    float* mbuf  = logit + (size_t)E * H;                         // N*H
    float* sbuf  = mbuf + (size_t)N * H;                          // N*H
    float* acc   = fd;
    float* hnext = fs;

    auto cdiv = [](int a, int b) { return (a + b - 1) / b; };

    for (int L = 0; L < 3; ++L) {
        const int Fin = FinArr[L], D = DArr[L], K = H * D;
        const float* hin = (L == 0) ? n_feat : hnext;

        k_cast_bf16<<<cdiv(N * Fin, 256), 256, 0, stream>>>(hin, Abf, N * Fin);
        k_pack_b<<<cdiv(Fin * K, 256), 256, 0, stream>>>(Wl[L], Plbf, Fin, K);
        k_pack_b<<<cdiv(Fin * K, 256), 256, 0, stream>>>(Wr[L], Prbf, Fin, K);

        dim3 g(cdiv(N / 16, 16), K / 32);
        k_gemm_wmma<<<g, 256, 0, stream>>>(Abf, Plbf, fs, N, Fin, K);
        k_gemm_wmma<<<g, 256, 0, stream>>>(Abf, Prbf, fd, N, Fin, K);

        k_fill<<<cdiv(N * H, 256), 256, 0, stream>>>(mbuf, -INFINITY, N * H);
        k_fill<<<cdiv(N * H, 256), 256, 0, stream>>>(sbuf, 0.f, N * H);
        k_edge_logit_max<<<cdiv(E * H, 256), 256, 0, stream>>>(fs, fd, src, dst, attn[L],
                                                               logit, mbuf, E, H, D);
        k_fill<<<cdiv(N * K, 256), 256, 0, stream>>>(acc, 0.f, N * K);  // fd dead now
        k_edge_exp_sum<<<cdiv(E * H, 256), 256, 0, stream>>>(logit, dst, mbuf, sbuf, E, H);
        k_edge_accum<<<cdiv(E * K, 256), 256, 0, stream>>>(logit, sbuf, fs, src, dst,
                                                           acc, E, H, D);
        k_finalize<<<cdiv(N * D, 256), 256, 0, stream>>>(acc, bias[L], hnext, N, H, D);
    }

    float* out = (float*)d_out;
    k_fill<<<1, 64, 0, stream>>>(out, 0.f, 64);
    k_readout<<<cdiv(N * 64, 256), 256, 0, stream>>>(hnext, out, N, 64);
}
